// HashingMemoryLite_51565377356296
// MI455X (gfx1250) — matrix operand order repaired and checked by reading, not compiled, not run on hardware
//
#include <hip/hip_runtime.h>
#include <hip/hip_bf16.h>

// ---------------- problem constants ----------------
constexpr int HEADS = 4;
constexpr int KDIM  = 512;
constexpr int HALF  = KDIM / 2;     // 256
constexpr int KNN   = 16;
constexpr int NKEYS = 512;
constexpr int INPUT_DIM = 1024;
constexpr int VDIM  = 512;
constexpr int BATCH = 2;
constexpr int TLEN  = 1024;
constexpr int BS    = BATCH * TLEN;     // 2048 rows
constexpr int QDIM  = HEADS * KDIM;     // 2048

// ---------------- WMMA types (CDNA5, wave32) ----------------
typedef __attribute__((ext_vector_type(16))) __bf16 v16bf;
typedef __attribute__((ext_vector_type(8)))  float  v8f;

union FragU { v16bf v; unsigned u[8]; };

// Padded LDS row: 32 K-elems bf16 = 16 dwords data, pad to 20 dwords (80B, 16B-aligned).
// Bank check: 20*i mod 64 is a permutation for i=0..15 -> conflict-free fragment reads.
constexpr int BROW = 20;
constexpr int PANEL = 64 * BROW;    // dwords per LDS B-panel buffer

// f32 -> bf16 round-to-nearest-even (bit pattern only)
__device__ __forceinline__ unsigned short f32_to_bf16(float f) {
  unsigned u = __float_as_uint(f);
  u += 0x7FFFu + ((u >> 16) & 1u);
  return (unsigned short)(u >> 16);
}

__global__ void __launch_bounds__(256) cvt_f32_to_bf16(const float* __restrict__ in,
                                                       unsigned short* __restrict__ out, int n) {
  int i = blockIdx.x * blockDim.x + threadIdx.x;
  if (i < n) out[i] = f32_to_bf16(in[i]);
}

// ---- A fragment (16x32 MxK, 16-bit layout per CDNA5 ISA §7.12.2) ----
// lane L: m = L&15 ; kgroup = (L>>4)*8 ; VGPR v holds K = (v<4?0:16)+kgroup+(v&3)*2 .. +1
__device__ __forceinline__ v16bf load_a_frag(const unsigned short* A, int lda,
                                             int m0, int k0, int lane) {
  const int m = m0 + (lane & 15);
  const int kg = (lane >> 4) * 8;
  const unsigned* p = (const unsigned*)(A + (size_t)m * lda + k0);
  FragU f;
#pragma unroll
  for (int v = 0; v < 8; ++v) {
    const int kb = ((v < 4) ? 0 : 16) + kg + (v & 3) * 2;
    f.u[v] = p[kb >> 1];
  }
  return f.v;
}

// ---- B fragment (32x16 KxN) read from the LDS-staged panel ----
// lane L: n = L&15 ; dword index (v<4?0:8) + (L>>4)*4 + (v&3) within the padded row
__device__ __forceinline__ v16bf load_b_frag_lds(const unsigned* lds, int nloc0, int lane) {
  const unsigned* p = lds + (nloc0 + (lane & 15)) * BROW + (lane >> 4) * 4;
  FragU f;
#pragma unroll
  for (int v = 0; v < 8; ++v) f.u[v] = p[((v < 4) ? 0 : 8) + (v & 3)];
  return f.v;
}

// ---- async stage of one 64N x 32K bf16 panel into LDS (ASYNCcnt path) ----
// 256 threads: thread t covers row t>>2, 16B chunk t&3.
__device__ __forceinline__ void stage_b_async(unsigned* lds, const unsigned short* Bt,
                                              int ldb, int n0, int k0, int tid) {
  const int row   = tid >> 2;
  const int chunk = tid & 3;
  // low 32 bits of a flat LDS pointer are the LDS byte address (aperture truncation)
  const unsigned lds_off = (unsigned)(size_t)(lds + row * BROW + chunk * 4);
  const unsigned short* g = Bt + (size_t)(n0 + row) * ldb + k0 + chunk * 8;
  asm volatile("global_load_async_to_lds_b128 %0, %1, off"
               :: "v"(lds_off), "v"(g) : "memory");
}

__device__ __forceinline__ void wait_async_all() {
  asm volatile("s_wait_asynccnt 0" ::: "memory");
}

// ---------------- shared GEMM tile body ----------------
// Block: 256 thr = 8 waves; block tile 128M x 64N; wave w owns M rows [w*16, w*16+16).
// B panel double-buffered in LDS via async copies; A fragments straight from global
// (each A row is loaded exactly once per block -> no cross-wave duplication).
// Manual 2x unroll with named buffers keeps accumulator PHIs in place (no v_mov chains),
// and all four B fragments are fetched before the WMMA chain so a single s_wait_dscnt
// covers the group and the 4 WMMAs issue back-to-back.
__device__ __forceinline__ void gemm_tile_body(const unsigned short* A, int lda,
                                               const unsigned short* Bt, int ldb,
                                               const float* bias,   // may be nullptr
                                               float* C, int ldc,
                                               int ksteps, unsigned* ldsB /*2*PANEL*/) {
  const int tid  = threadIdx.x;
  const int lane = tid & 31;
  const int wave = tid >> 5;
  const int m0 = blockIdx.y * 128 + wave * 16;
  const int n0 = blockIdx.x * 64;
  unsigned* buf0 = ldsB;
  unsigned* buf1 = ldsB + PANEL;

  v8f acc0 = v8f{}, acc1 = v8f{}, acc2 = v8f{}, acc3 = v8f{};

  stage_b_async(buf0, Bt, ldb, n0, 0, tid);

#pragma unroll 1
  for (int s = 0; s < ksteps; s += 2) {
    // ---- even step: compute from buf0, prefetch into buf1 ----
    wait_async_all();
    __syncthreads();
    if (s + 1 < ksteps) stage_b_async(buf1, Bt, ldb, n0, (s + 1) * 32, tid);
    {
      v16bf a  = load_a_frag(A, lda, m0, s * 32, lane);
      v16bf b0 = load_b_frag_lds(buf0,  0, lane);
      v16bf b1 = load_b_frag_lds(buf0, 16, lane);
      v16bf b2 = load_b_frag_lds(buf0, 32, lane);
      v16bf b3 = load_b_frag_lds(buf0, 48, lane);
      acc0 = __builtin_amdgcn_wmma_f32_16x16x32_bf16(false, a, false, b0, (short)0, acc0, false, false);
      acc1 = __builtin_amdgcn_wmma_f32_16x16x32_bf16(false, a, false, b1, (short)0, acc1, false, false);
      acc2 = __builtin_amdgcn_wmma_f32_16x16x32_bf16(false, a, false, b2, (short)0, acc2, false, false);
      acc3 = __builtin_amdgcn_wmma_f32_16x16x32_bf16(false, a, false, b3, (short)0, acc3, false, false);
    }
    if (s + 1 >= ksteps) break;
    // ---- odd step: compute from buf1, prefetch into buf0 ----
    wait_async_all();
    __syncthreads();
    if (s + 2 < ksteps) stage_b_async(buf0, Bt, ldb, n0, (s + 2) * 32, tid);
    {
      v16bf a  = load_a_frag(A, lda, m0, (s + 1) * 32, lane);
      v16bf b0 = load_b_frag_lds(buf1,  0, lane);
      v16bf b1 = load_b_frag_lds(buf1, 16, lane);
      v16bf b2 = load_b_frag_lds(buf1, 32, lane);
      v16bf b3 = load_b_frag_lds(buf1, 48, lane);
      acc0 = __builtin_amdgcn_wmma_f32_16x16x32_bf16(false, a, false, b0, (short)0, acc0, false, false);
      acc1 = __builtin_amdgcn_wmma_f32_16x16x32_bf16(false, a, false, b1, (short)0, acc1, false, false);
      acc2 = __builtin_amdgcn_wmma_f32_16x16x32_bf16(false, a, false, b2, (short)0, acc2, false, false);
      acc3 = __builtin_amdgcn_wmma_f32_16x16x32_bf16(false, a, false, b3, (short)0, acc3, false, false);
    }
  }

  // D layout: lane L -> n = L&15 ; VGPR r -> m = (L>>4)*8 + r
  const int mrow = m0 + (lane >> 4) * 8;
  const int ncol = lane & 15;
  v8f accs[4] = {acc0, acc1, acc2, acc3};
#pragma unroll
  for (int t = 0; t < 4; ++t) {
    const int n = n0 + t * 16 + ncol;
    const float bb = bias ? bias[n] : 0.0f;
#pragma unroll
    for (int r = 0; r < 8; ++r)
      C[(size_t)(mrow + r) * ldc + n] = accs[t][r] + bb;
  }
}

// ---------------- kernel 1: q = x @ Wq^T + bq ----------------
__global__ void __launch_bounds__(256) gemm_q_kernel(const unsigned short* __restrict__ xb,
                                                     const unsigned short* __restrict__ wqb,
                                                     const float* __restrict__ bq,
                                                     float* __restrict__ q) {
  __shared__ unsigned ldsB[2 * PANEL];
  gemm_tile_body(xb, INPUT_DIM, wqb, INPUT_DIM, bq, q, QDIM, INPUT_DIM / 32, ldsB);
}

// ---------------- kernel 2: scores s[j][h] = q_half @ k_half^T ----------------
__global__ void __launch_bounds__(256) gemm_scores_kernel(const unsigned short* __restrict__ qb,
                                                          const unsigned short* __restrict__ keysb,
                                                          float* __restrict__ sbuf) {
  __shared__ unsigned ldsB[2 * PANEL];
  const int z = blockIdx.z;
  const int j = z >> 2;            // half 0/1
  const int h = z & 3;             // head
  const unsigned short* A  = qb + h * KDIM + j * HALF;                   // stride QDIM
  const unsigned short* Bt = keysb + (size_t)(h * 2 + j) * NKEYS * HALF; // stride HALF
  float* S = sbuf + (size_t)z * BS * NKEYS;
  gemm_tile_body(A, QDIM, Bt, HALF, nullptr, S, NKEYS, HALF / 32, ldsB);
}

// ---------------- kernel 3: top-16 of 512 per row (one wave32 per row) ----------------
__global__ void __launch_bounds__(256) topk512_kernel(const float* __restrict__ sbuf,
                                                      float* __restrict__ tS,
                                                      int* __restrict__ tI) {
  const int row  = blockIdx.x * 8 + (threadIdx.x >> 5);
  const int lane = threadIdx.x & 31;
  const float* s = sbuf + (size_t)row * NKEYS;
  float v[16];
#pragma unroll
  for (int i = 0; i < 16; ++i) v[i] = s[i * 32 + lane];   // element n = i*32 + lane
  float* oS = tS + (size_t)row * KNN;
  int*   oI = tI + (size_t)row * KNN;
  for (int t = 0; t < KNN; ++t) {
    float bv = -__builtin_inff(); int bi = 0;
#pragma unroll
    for (int i = 0; i < 16; ++i) if (v[i] > bv) { bv = v[i]; bi = i; }
    float rv = bv; int ri = bi * 32 + lane;
    for (int off = 16; off; off >>= 1) {
      float ov = __shfl_down(rv, off, 32);
      int   oi = __shfl_down(ri, off, 32);
      if (ov > rv || (ov == rv && oi < ri)) { rv = ov; ri = oi; }
    }
    rv = __shfl(rv, 0, 32); ri = __shfl(ri, 0, 32);
    if (lane == (ri & 31)) {
      const int slot = ri >> 5;
#pragma unroll
      for (int i = 0; i < 16; ++i) if (i == slot) v[i] = -__builtin_inff();
    }
    if (lane == 0) { oS[t] = rv; oI[t] = ri; }
  }
}

// ---------------- kernel 4: combine 16x16 candidates, top-16, softmax ----------------
__global__ void __launch_bounds__(256) combine_kernel(const float* __restrict__ tS,
                                                      const int* __restrict__ tI,
                                                      float* __restrict__ wf,
                                                      int* __restrict__ idxf) {
  const int row  = blockIdx.x * 8 + (threadIdx.x >> 5);  // row = b*HEADS + h
  const int lane = threadIdx.x & 31;
  const int b = row >> 2, h = row & 3;
  const float* s1 = tS + ((size_t)(0 * HEADS + h) * BS + b) * KNN;
  const float* s2 = tS + ((size_t)(1 * HEADS + h) * BS + b) * KNN;
  const int*   i1 = tI + ((size_t)(0 * HEADS + h) * BS + b) * KNN;
  const int*   i2 = tI + ((size_t)(1 * HEADS + h) * BS + b) * KNN;
  float v[8];
#pragma unroll
  for (int c = 0; c < 8; ++c) {
    const int cand = lane * 8 + c;                       // cand = a*16 + bb
    v[c] = s1[cand >> 4] + s2[cand & 15];
  }
  float selV = -__builtin_inff(); int selC = 0;
  for (int t = 0; t < KNN; ++t) {
    float bv = -__builtin_inff(); int bi = 0;
#pragma unroll
    for (int i = 0; i < 8; ++i) if (v[i] > bv) { bv = v[i]; bi = i; }
    float rv = bv; int ri = lane * 8 + bi;
    for (int off = 16; off; off >>= 1) {
      float ov = __shfl_down(rv, off, 32);
      int   oi = __shfl_down(ri, off, 32);
      if (ov > rv || (ov == rv && oi < ri)) { rv = ov; ri = oi; }
    }
    rv = __shfl(rv, 0, 32); ri = __shfl(ri, 0, 32);
    if (lane == (ri >> 3)) {
      const int slot = ri & 7;
#pragma unroll
      for (int i = 0; i < 8; ++i) if (i == slot) v[i] = -__builtin_inff();
    }
    if (lane == t) { selV = rv; selC = ri; }             // lane t owns t-th selection
  }
  // softmax across lanes 0..15 (width-16 xor shuffles)
  float mx = selV;
  for (int off = 8; off; off >>= 1) mx = fmaxf(mx, __shfl_xor(mx, off, 16));
  float e = __expf(selV - mx);
  float sum = e;
  for (int off = 8; off; off >>= 1) sum += __shfl_xor(sum, off, 16);
  if (lane < KNN) {
    const int c = selC;
    wf[(size_t)row * KNN + lane]   = e / sum;
    idxf[(size_t)row * KNN + lane] = i1[c >> 4] * NKEYS + i2[c & 15];
  }
}

// ---------------- kernel 5: out[b] = sum_k w[k] * values[idx[k]] ----------------
__global__ void __launch_bounds__(128) gather_kernel(const float* __restrict__ wf,
                                                     const int* __restrict__ idxf,
                                                     const float* __restrict__ values,
                                                     float* __restrict__ out) {
  const int b = blockIdx.x;
  const int t = threadIdx.x;                 // 128 threads, 4 columns each
  __shared__ float sw[HEADS * KNN];
  __shared__ int   si[HEADS * KNN];
  if (t < HEADS * KNN) {
    sw[t] = wf[(size_t)b * HEADS * KNN + t];
    si[t] = idxf[(size_t)b * HEADS * KNN + t];
  }
  __syncthreads();
  const int col = t * 4;
  float4 acc = make_float4(0.f, 0.f, 0.f, 0.f);
  for (int k = 0; k < HEADS * KNN; ++k) {
    const float  w  = sw[k];
    const float4 vv = *(const float4*)(values + (size_t)si[k] * VDIM + col);
    acc.x += w * vv.x; acc.y += w * vv.y; acc.z += w * vv.z; acc.w += w * vv.w;
  }
  *(float4*)(out + (size_t)b * VDIM + col) = acc;
}

// ---------------- host-side orchestration ----------------
extern "C" void kernel_launch(void* const* d_in, const int* in_sizes, int n_in,
                              void* d_out, int out_size, void* d_ws, size_t ws_size,
                              hipStream_t stream) {
  (void)in_sizes; (void)n_in; (void)out_size; (void)ws_size;
  const float* x      = (const float*)d_in[0];
  const float* Wq     = (const float*)d_in[1];
  const float* bq     = (const float*)d_in[2];
  const float* keys   = (const float*)d_in[3];
  const float* values = (const float*)d_in[4];
  float* out = (float*)d_out;

  // workspace carve-up (all regions fully overwritten before use)
  char* w = (char*)d_ws;
  auto carve = [&](size_t bytes) { void* p = (void*)w; w += (bytes + 255) & ~(size_t)255; return p; };
  unsigned short* xb    = (unsigned short*)carve((size_t)BS * INPUT_DIM * 2);
  unsigned short* wqb   = (unsigned short*)carve((size_t)QDIM * INPUT_DIM * 2);
  unsigned short* keysb = (unsigned short*)carve((size_t)2 * HEADS * NKEYS * HALF * 2);
  float*          qf    = (float*)carve((size_t)BS * QDIM * 4);
  unsigned short* qb    = (unsigned short*)carve((size_t)BS * QDIM * 2);
  float*          sbuf  = (float*)carve((size_t)2 * HEADS * BS * NKEYS * 4);
  float*          tS    = (float*)carve((size_t)2 * HEADS * BS * KNN * 4);
  int*            tI    = (int*)carve((size_t)2 * HEADS * BS * KNN * 4);
  float*          wfb   = (float*)carve((size_t)BS * HEADS * KNN * 4);
  int*            idxf  = (int*)carve((size_t)BS * HEADS * KNN * 4);

  // 1) fp32 -> bf16 conversions (working set then lives in 192MB L2)
  {
    int n1 = BS * INPUT_DIM;
    cvt_f32_to_bf16<<<(n1 + 255) / 256, 256, 0, stream>>>(x, xb, n1);
    int n2 = QDIM * INPUT_DIM;
    cvt_f32_to_bf16<<<(n2 + 255) / 256, 256, 0, stream>>>(Wq, wqb, n2);
    int n3 = 2 * HEADS * NKEYS * HALF;
    cvt_f32_to_bf16<<<(n3 + 255) / 256, 256, 0, stream>>>(keys, keysb, n3);
  }
  // 2) q projection (bf16 WMMA, async-LDS B panel, fused bias)
  gemm_q_kernel<<<dim3(QDIM / 64, BS / 128, 1), 256, 0, stream>>>(xb, wqb, bq, qf);
  // 3) q -> bf16 for score GEMMs
  {
    int n4 = BS * QDIM;
    cvt_f32_to_bf16<<<(n4 + 255) / 256, 256, 0, stream>>>(qf, qb, n4);
  }
  // 4) per-(half,head) score GEMMs
  gemm_scores_kernel<<<dim3(NKEYS / 64, BS / 128, 2 * HEADS), 256, 0, stream>>>(qb, keysb, sbuf);
  // 5) top-16 over 512 keys, per (half, head, row)
  topk512_kernel<<<(2 * HEADS * BS) / 8, 256, 0, stream>>>(sbuf, tS, tI);
  // 6) combine halves, top-16 of 256 candidates, softmax
  combine_kernel<<<(BS * HEADS) / 8, 256, 0, stream>>>(tS, tI, wfb, idxf);
  // 7) weighted gather from values
  gather_kernel<<<BS, 128, 0, stream>>>(wfb, idxf, values, out);
}
